// MLAAttention_84928683311386
// MI455X (gfx1250) — compile-verified, matrix operand order checked by reading, not run on hardware
//
#include <hip/hip_runtime.h>
#include <hip/hip_bf16.h>
#include <cstdint>

// ---------------------------------------------------------------------------
// MLA attention forward for MI455X (gfx1250, wave32, WMMA, async LDS copies).
// ---------------------------------------------------------------------------

typedef __attribute__((ext_vector_type(16))) __bf16 bf16x16;
typedef __attribute__((ext_vector_type(8)))  __bf16 bf16x8;
typedef __attribute__((ext_vector_type(8)))  float  floatx8;
typedef __attribute__((ext_vector_type(4)))  float  floatx4;
typedef __attribute__((ext_vector_type(4)))  int    intx4;
typedef __attribute__((ext_vector_type(4)))  unsigned int uintx4;

#if defined(__has_builtin)
#if __has_builtin(__builtin_amdgcn_global_load_async_to_lds_b128)
#define HAVE_ASYNC_LDS 1
#endif
#endif
#ifndef HAVE_ASYNC_LDS
#define HAVE_ASYNC_LDS 0
#endif

#if HAVE_ASYNC_LDS
// Probe-confirmed signature: (int4 AS1* src, int4 AS3* dst, imm offset, imm cpol)
typedef intx4 __attribute__((address_space(1)))* g4ptr_t;
typedef intx4 __attribute__((address_space(3)))* l4ptr_t;
#endif

// Copy 16 bytes global -> LDS. Async (ASYNCcnt-tracked) when available.
static __device__ __forceinline__ void async_copy16(const void* g, void* l) {
#if HAVE_ASYNC_LDS
  __builtin_amdgcn_global_load_async_to_lds_b128(
      (g4ptr_t)(uintptr_t)g,
      (l4ptr_t)(unsigned)(uintptr_t)l, 0, 0);
#else
  *(uintx4*)l = *(const uintx4*)g;
#endif
}

static __device__ __forceinline__ void async_wait0() {
#if HAVE_ASYNC_LDS
#if __has_builtin(__builtin_amdgcn_s_wait_asynccnt)
  __builtin_amdgcn_s_wait_asynccnt(0);
#else
  asm volatile("s_wait_asynccnt 0x0" ::: "memory");
#endif
#endif
}

static __device__ __forceinline__ floatx8 wmma_bf16(bf16x16 a, bf16x16 b, floatx8 c) {
  return __builtin_amdgcn_wmma_f32_16x16x32_bf16(false, a, false, b, (short)0, c, false, false);
}

// 16x32 bf16 fragment from LDS (ISA 7.12.2 16-bit A layout); stride%8==0.
static __device__ __forceinline__ bf16x16 load_frag(const __bf16* base, int stride, int lane) {
  int r  = lane & 15;
  int kh = (lane >> 4) & 1;
  const __bf16* p = base + r * stride + kh * 8;
  bf16x8 lo = *(const bf16x8*)(p);
  bf16x8 hi = *(const bf16x8*)(p + 16);
  return __builtin_shufflevector(lo, hi, 0,1,2,3,4,5,6,7,8,9,10,11,12,13,14,15);
}

// ---------------------------------------------------------------------------
// Elementwise f32 -> bf16 (8 elems/thread, b128 loads, b128 store).
// ---------------------------------------------------------------------------
__global__ __launch_bounds__(256) void conv_f32_to_bf16(
    const float* __restrict__ x, __bf16* __restrict__ y, long long n) {
  long long i = ((long long)blockIdx.x * 256 + threadIdx.x) * 8;
  if (i + 8 <= n) {
    floatx4 a = *(const floatx4*)(x + i);
    floatx4 b = *(const floatx4*)(x + i + 4);
    bf16x8 o;
    #pragma unroll
    for (int j = 0; j < 4; ++j) { o[j] = (__bf16)a[j]; o[j + 4] = (__bf16)b[j]; }
    *(bf16x8*)(y + i) = o;
  }
}

// ---------------------------------------------------------------------------
// Weight transpose + convert: W[K,N] f32 -> Wt[N,K] bf16 (32x32 LDS tiles).
// ---------------------------------------------------------------------------
__global__ __launch_bounds__(256) void wtrans_kernel(
    const float* __restrict__ W, __bf16* __restrict__ Wt, int K, int N) {
  __shared__ float tile[32][33];
  const int n0 = blockIdx.x * 32, k0 = blockIdx.y * 32;
  const int t = threadIdx.x;
  #pragma unroll
  for (int i = 0; i < 4; ++i) {
    int e = i * 256 + t; int r = e >> 5, c = e & 31;
    tile[r][c] = W[(size_t)(k0 + r) * N + n0 + c];
  }
  __syncthreads();
  #pragma unroll
  for (int i = 0; i < 4; ++i) {
    int e = i * 256 + t; int r = e >> 5, c = e & 31;
    Wt[(size_t)(n0 + r) * K + k0 + c] = (__bf16)tile[c][r];
  }
}

// V transpose: kv_bf[token, h*256+128+n] -> vt[(b,h,n), key] (bf16).
__global__ __launch_bounds__(256) void vtrans_kernel(
    const __bf16* __restrict__ kvb, __bf16* __restrict__ vt) {
  __shared__ __bf16 tile[32][36];
  const int s0 = blockIdx.x * 32, n0 = blockIdx.y * 32, bh = blockIdx.z;
  const int b = bh >> 4, h = bh & 15;
  const int t = threadIdx.x;
  #pragma unroll
  for (int i = 0; i < 4; ++i) {
    int e = i * 256 + t; int r = e >> 5, c = e & 31;
    tile[r][c] = kvb[(size_t)(b * 2048 + s0 + r) * 4096 + h * 256 + 128 + n0 + c];
  }
  __syncthreads();
  #pragma unroll
  for (int i = 0; i < 4; ++i) {
    int e = i * 256 + t; int r = e >> 5, c = e & 31;
    vt[((size_t)bh * 128 + n0 + r) * 2048 + s0 + c] = tile[c][r];
  }
}

// ---------------------------------------------------------------------------
// GEMM: C[M,N] = A[M,K] * Bt[N,K]^T, bf16 inputs, f32 accumulate.
// Tile 128x64xBK32, 256 threads = 8 waves (32x32 each), double-buffered LDS
// filled with 16B async copies. Optional f32 and/or bf16 outputs.
// ---------------------------------------------------------------------------
#define GM 128
#define GN 64
#define GK 32
#define ASTR 40  // 32 + 8 pad; rows are 80B (16B-aligned)

__global__ __launch_bounds__(256) void gemm_bf16_wmma(
    const __bf16* __restrict__ A, const __bf16* __restrict__ Bt,
    float* __restrict__ Cf, __bf16* __restrict__ Cb, int M, int N, int K) {
  __shared__ __attribute__((aligned(16))) __bf16 As[2][GM * ASTR];
  __shared__ __attribute__((aligned(16))) __bf16 Bs[2][GN * ASTR];

  const int m0 = blockIdx.y * GM;
  const int n0 = blockIdx.x * GN;
  const int t  = threadIdx.x;
  const int lane = t & 31;
  const int w    = t >> 5;
  const int mbase = (w >> 1) * 32;
  const int nbase = (w & 1) * 32;

  auto stage = [&](int buf, int k0) {
    // A tile: 128 rows x 64B -> 512 x 16B chunks, 2 per thread.
    #pragma unroll
    for (int i = 0; i < 2; ++i) {
      int e = i * 256 + t;
      int r = e >> 2, c = e & 3;
      async_copy16(A + (size_t)(m0 + r) * K + k0 + c * 8,
                   &As[buf][r * ASTR + c * 8]);
    }
    // B tile: 64 rows x 64B -> 256 x 16B chunks, 1 per thread.
    {
      int r = t >> 2, c = t & 3;
      async_copy16(Bt + (size_t)(n0 + r) * K + k0 + c * 8,
                   &Bs[buf][r * ASTR + c * 8]);
    }
  };

  floatx8 acc00 = {0,0,0,0,0,0,0,0};
  floatx8 acc01 = {0,0,0,0,0,0,0,0};
  floatx8 acc10 = {0,0,0,0,0,0,0,0};
  floatx8 acc11 = {0,0,0,0,0,0,0,0};

  stage(0, 0);
  const int nk = K / GK;
  for (int ki = 0; ki < nk; ++ki) {
    const int buf = ki & 1;
    async_wait0();
    __syncthreads();                       // tile `buf` visible to all waves
    if (ki + 1 < nk) stage(buf ^ 1, (ki + 1) * GK);  // overlap next copy

    bf16x16 a0 = load_frag(&As[buf][mbase * ASTR],        ASTR, lane);
    bf16x16 a1 = load_frag(&As[buf][(mbase + 16) * ASTR], ASTR, lane);
    bf16x16 b0 = load_frag(&Bs[buf][nbase * ASTR],        ASTR, lane);
    bf16x16 b1 = load_frag(&Bs[buf][(nbase + 16) * ASTR], ASTR, lane);
    acc00 = wmma_bf16(a0, b0, acc00);
    acc01 = wmma_bf16(a0, b1, acc01);
    acc10 = wmma_bf16(a1, b0, acc10);
    acc11 = wmma_bf16(a1, b1, acc11);
  }

  const int rsub = (lane >> 4) * 8;
  const int csub = lane & 15;
  #pragma unroll
  for (int j = 0; j < 8; ++j) {
    int m = m0 + mbase + rsub + j;
    int n = n0 + nbase + csub;
    if (Cf) {
      Cf[(size_t)m * N + n]             = acc00[j];
      Cf[(size_t)m * N + n + 16]        = acc01[j];
      Cf[(size_t)(m + 16) * N + n]      = acc10[j];
      Cf[(size_t)(m + 16) * N + n + 16] = acc11[j];
    }
    if (Cb) {
      Cb[(size_t)m * N + n]             = (__bf16)acc00[j];
      Cb[(size_t)m * N + n + 16]        = (__bf16)acc01[j];
      Cb[(size_t)(m + 16) * N + n]      = (__bf16)acc10[j];
      Cb[(size_t)(m + 16) * N + n + 16] = (__bf16)acc11[j];
    }
  }
}

// ---------------------------------------------------------------------------
// RMSNorm over rows, f32 in, bf16 out (feeds the next GEMM's A operand).
// ---------------------------------------------------------------------------
__global__ __launch_bounds__(256) void rmsnorm_kernel(
    const float* __restrict__ x, const float* __restrict__ w,
    __bf16* __restrict__ y, int cols, int in_stride, int out_stride) {
  const int row = blockIdx.x;
  const float* xr = x + (size_t)row * in_stride;
  __bf16* yr = y + (size_t)row * out_stride;
  float ss = 0.f;
  for (int c = threadIdx.x; c < cols; c += 256) { float v = xr[c]; ss += v * v; }
  __shared__ float red[256];
  red[threadIdx.x] = ss;
  __syncthreads();
  for (int s = 128; s > 0; s >>= 1) {
    if (threadIdx.x < (unsigned)s) red[threadIdx.x] += red[threadIdx.x + s];
    __syncthreads();
  }
  const float scale = rsqrtf(red[0] / (float)cols + 1e-6f);
  for (int c = threadIdx.x; c < cols; c += 256) yr[c] = (__bf16)(xr[c] * scale * w[c]);
}

// ---------------------------------------------------------------------------
// RoPE on q rope dims: read f32 qbuf, write bf16 into q_bf (nope dims were
// already written by the dual-output GEMM).
// ---------------------------------------------------------------------------
__global__ __launch_bounds__(512) void rope_q_kernel(
    const float* __restrict__ qf, __bf16* __restrict__ qb,
    const int* __restrict__ positions) {
  const int token = blockIdx.x;
  const int t = threadIdx.x;
  const int h = t >> 5, i = t & 31;
  const float pos = (float)positions[token];
  const float inv = __powf(10000.0f, -(float)(2 * i) / 64.0f);
  float s, c;
  __sincosf(pos * inv, &s, &c);
  const float* bi = qf + (size_t)token * 3072 + h * 192 + 128;
  __bf16* bo = qb + (size_t)token * 3072 + h * 192 + 128;
  const float x1 = bi[i], x2 = bi[32 + i];
  bo[i]      = (__bf16)(x1 * c - x2 * s);
  bo[32 + i] = (__bf16)(x2 * c + x1 * s);
}

// RoPE on shared k_rope: kv_comb[:,512:576] f32 -> kr bf16 [B*S,64].
__global__ __launch_bounds__(256) void rope_k_kernel(
    const float* __restrict__ kvc, const int* __restrict__ positions,
    __bf16* __restrict__ kr) {
  const int idx = blockIdx.x * 256 + threadIdx.x;
  const int token = idx >> 5, i = idx & 31;
  const float pos = (float)positions[token];
  const float inv = __powf(10000.0f, -(float)(2 * i) / 64.0f);
  float s, c;
  __sincosf(pos * inv, &s, &c);
  const float* src = kvc + (size_t)token * 576 + 512;
  const float x1 = src[i], x2 = src[32 + i];
  kr[(size_t)token * 64 + i]      = (__bf16)(x1 * c - x2 * s);
  kr[(size_t)token * 64 + 32 + i] = (__bf16)(x2 * c + x1 * s);
}

// ---------------------------------------------------------------------------
// Flash-style causal MLA attention: block = (b, h, 64-row q tile), 32-key
// K tiles. All staging is 16B async copies from pre-packed bf16 buffers.
// Output bf16 (feeds the final GEMM).
// ---------------------------------------------------------------------------
#define QSTRD 200
#define VSTRD 40
#define SSTRD 36

__global__ __launch_bounds__(256) void mla_attn_kernel(
    const __bf16* __restrict__ q, const __bf16* __restrict__ kv,
    const __bf16* __restrict__ kr, const __bf16* __restrict__ vt,
    __bf16* __restrict__ out) {
  const int qt = blockIdx.x;
  const int h  = blockIdx.y;
  const int b  = blockIdx.z;
  const int q0 = qt * 64;
  const size_t tok0 = (size_t)b * 2048;
  const size_t vbase = ((size_t)(b * 16 + h)) * 128 * 2048;

  __shared__ __attribute__((aligned(16))) __bf16 Ql[64 * QSTRD];
  __shared__ __attribute__((aligned(16))) __bf16 Kl[32 * QSTRD];
  __shared__ __attribute__((aligned(16))) __bf16 Vt[128 * VSTRD];
  __shared__ __attribute__((aligned(16))) __bf16 Pl[64 * VSTRD];
  __shared__ float Sl[64 * SSTRD];
  __shared__ float alpha_s[64];
  __shared__ float lrow[64];

  const int t = threadIdx.x;
  const int lane = t & 31;
  const int w = t >> 5;
  const int mt   = w & 3;
  const int half = w >> 2;

  // Q tile 64x192: 24 chunks/row -> 1536 chunks, 6 per thread.
  #pragma unroll
  for (int i = 0; i < 6; ++i) {
    int e = i * 256 + t;
    int r = e / 24, c = e % 24;
    async_copy16(q + (tok0 + q0 + r) * 3072 + (size_t)h * 192 + c * 8,
                 &Ql[r * QSTRD + c * 8]);
  }

  float m_i = -1e30f, l_i = 0.f;
  floatx8 o0 = {0,0,0,0,0,0,0,0};
  floatx8 o1 = {0,0,0,0,0,0,0,0};
  floatx8 o2 = {0,0,0,0,0,0,0,0};
  floatx8 o3 = {0,0,0,0,0,0,0,0};

  const int nkt = (q0 + 64) / 32;
  for (int kt = 0; kt < nkt; ++kt) {
    const int k0 = kt * 32;
    __syncthreads();  // prior reads of Kl/Vt complete before re-staging

    // K tile 32x192: nope (16 chunks) + rope (8 chunks) per row; 3/thread.
    #pragma unroll
    for (int i = 0; i < 3; ++i) {
      int e = i * 256 + t;
      int r = e / 24, c = e % 24;
      const __bf16* g = (c < 16)
          ? kv + (tok0 + k0 + r) * 4096 + (size_t)h * 256 + c * 8
          : kr + (tok0 + k0 + r) * 64 + (c - 16) * 8;
      async_copy16(g, &Kl[r * QSTRD + c * 8]);
    }
    // V tile transposed source: 128 dims x 64B -> 512 chunks, 2/thread.
    #pragma unroll
    for (int i = 0; i < 2; ++i) {
      int e = i * 256 + t;
      int r = e >> 2, c = e & 3;
      async_copy16(vt + vbase + (size_t)r * 2048 + k0 + c * 8,
                   &Vt[r * VSTRD + c * 8]);
    }
    async_wait0();
    __syncthreads();

    // Scores 64x32 = Q * K^T over d=192 (6 WMMA per wave).
    {
      floatx8 sacc = {0,0,0,0,0,0,0,0};
      #pragma unroll
      for (int kc = 0; kc < 6; ++kc) {
        bf16x16 a  = load_frag(&Ql[(mt * 16) * QSTRD + kc * 32], QSTRD, lane);
        bf16x16 bb = load_frag(&Kl[(half * 16) * QSTRD + kc * 32], QSTRD, lane);
        sacc = wmma_bf16(a, bb, sacc);
      }
      const int rsub = (lane >> 4) * 8, csub = lane & 15;
      #pragma unroll
      for (int j = 0; j < 8; ++j)
        Sl[(mt * 16 + rsub + j) * SSTRD + half * 16 + csub] =
            sacc[j] * 0.07216878364870323f;  // 192^-0.5
    }
    __syncthreads();

    // Online softmax row pass (thread t<64 owns row t).
    if (t < 64) {
      const int qglob = q0 + t;
      float sv[32];
      float tmax = -1e30f;
      #pragma unroll
      for (int c = 0; c < 32; ++c) {
        float v = Sl[t * SSTRD + c];
        v = (k0 + c <= qglob) ? v : -1e30f;  // causal mask
        sv[c] = v;
        tmax = fmaxf(tmax, v);
      }
      const float newm  = fmaxf(m_i, tmax);
      const float alpha = __expf(m_i - newm);
      float psum = 0.f;
      #pragma unroll
      for (int c = 0; c < 32; ++c) {
        float p = (sv[c] > -1e29f) ? __expf(sv[c] - newm) : 0.f;
        psum += p;
        Pl[t * VSTRD + c] = (__bf16)p;
      }
      l_i = l_i * alpha + psum;
      m_i = newm;
      alpha_s[t] = alpha;
    }
    __syncthreads();

    // Rescale O, accumulate P * V (4 WMMA per wave).
    {
      const int rsub = (lane >> 4) * 8;
      bf16x16 pf = load_frag(&Pl[(mt * 16) * VSTRD], VSTRD, lane);
      bf16x16 v0 = load_frag(&Vt[(half * 64 +  0) * VSTRD], VSTRD, lane);
      bf16x16 v1 = load_frag(&Vt[(half * 64 + 16) * VSTRD], VSTRD, lane);
      bf16x16 v2 = load_frag(&Vt[(half * 64 + 32) * VSTRD], VSTRD, lane);
      bf16x16 v3 = load_frag(&Vt[(half * 64 + 48) * VSTRD], VSTRD, lane);
      float al[8];
      #pragma unroll
      for (int j = 0; j < 8; ++j) al[j] = alpha_s[mt * 16 + rsub + j];
      #pragma unroll
      for (int j = 0; j < 8; ++j) {
        o0[j] *= al[j]; o1[j] *= al[j]; o2[j] *= al[j]; o3[j] *= al[j];
      }
      o0 = wmma_bf16(pf, v0, o0);
      o1 = wmma_bf16(pf, v1, o1);
      o2 = wmma_bf16(pf, v2, o2);
      o3 = wmma_bf16(pf, v3, o3);
    }
  }

  if (t < 64) lrow[t] = l_i;
  __syncthreads();

  {
    const int rsub = (lane >> 4) * 8, csub = lane & 15;
    #pragma unroll
    for (int j = 0; j < 8; ++j) {
      const int row = mt * 16 + rsub + j;
      const float inv = 1.0f / lrow[row];
      const size_t obase =
          (tok0 + q0 + row) * 2048 + (size_t)h * 128 + half * 64 + csub;
      out[obase +  0] = (__bf16)(o0[j] * inv);
      out[obase + 16] = (__bf16)(o1[j] * inv);
      out[obase + 32] = (__bf16)(o2[j] * inv);
      out[obase + 48] = (__bf16)(o3[j] * inv);
    }
  }
}

// ---------------------------------------------------------------------------
// Host-side pipeline.
// ---------------------------------------------------------------------------
extern "C" void kernel_launch(void* const* d_in, const int* in_sizes, int n_in,
                              void* d_out, int out_size, void* d_ws, size_t ws_size,
                              hipStream_t stream) {
  (void)in_sizes; (void)n_in; (void)out_size; (void)ws_size;
  const float* hsrc      = (const float*)d_in[0];
  const int*   positions = (const int*)d_in[1];
  // d_in[2] attention_mask == causal 0/-1e9: applied analytically in-kernel.
  const float* w_q_a     = (const float*)d_in[3];
  const float* q_a_ln_w  = (const float*)d_in[4];
  const float* w_q_b     = (const float*)d_in[5];
  const float* w_kv_a    = (const float*)d_in[6];
  const float* kv_a_ln_w = (const float*)d_in[7];
  const float* w_kv_b    = (const float*)d_in[8];
  const float* w_o       = (const float*)d_in[9];
  float* out = (float*)d_out;

  char* base = (char*)d_ws;
  size_t off = 0;
  auto alloc = [&](size_t bytes) -> void* {
    void* p = base + off;
    off = (off + bytes + 255) & ~(size_t)255;
    return p;
  };
  // f32 intermediates
  float* q_pre   = (float*)alloc((size_t)4096 * 1536 * 4);
  float* qbuf    = (float*)alloc((size_t)4096 * 3072 * 4);
  float* kv_comb = (float*)alloc((size_t)4096 * 576 * 4);
  // bf16 operands
  __bf16* h_bf    = (__bf16*)alloc((size_t)4096 * 2048 * 2);
  __bf16* wqa_t   = (__bf16*)alloc((size_t)1536 * 2048 * 2);
  __bf16* wqb_t   = (__bf16*)alloc((size_t)3072 * 1536 * 2);
  __bf16* wkva_t  = (__bf16*)alloc((size_t)576 * 2048 * 2);
  __bf16* wkvb_t  = (__bf16*)alloc((size_t)4096 * 512 * 2);
  __bf16* wo_t    = (__bf16*)alloc((size_t)2048 * 2048 * 2);
  __bf16* q_c_bf  = (__bf16*)alloc((size_t)4096 * 1536 * 2);
  __bf16* kv_c_bf = (__bf16*)alloc((size_t)4096 * 512 * 2);
  __bf16* q_bf    = (__bf16*)alloc((size_t)4096 * 3072 * 2);
  __bf16* kv_bf   = (__bf16*)alloc((size_t)4096 * 4096 * 2);
  __bf16* vt_bf   = (__bf16*)alloc((size_t)32 * 128 * 2048 * 2);
  __bf16* kr_bf   = (__bf16*)alloc((size_t)4096 * 64 * 2);
  __bf16* attn_bf = (__bf16*)alloc((size_t)4096 * 2048 * 2);

  // Pack operands to bf16 (weights transposed to [N,K]).
  conv_f32_to_bf16<<<4096, 256, 0, stream>>>(hsrc, h_bf, (long long)4096 * 2048);
  wtrans_kernel<<<dim3(1536 / 32, 2048 / 32), 256, 0, stream>>>(w_q_a,  wqa_t,  2048, 1536);
  wtrans_kernel<<<dim3(3072 / 32, 1536 / 32), 256, 0, stream>>>(w_q_b,  wqb_t,  1536, 3072);
  wtrans_kernel<<<dim3(576  / 32, 2048 / 32), 256, 0, stream>>>(w_kv_a, wkva_t, 2048, 576);
  wtrans_kernel<<<dim3(4096 / 32, 512  / 32), 256, 0, stream>>>(w_kv_b, wkvb_t, 512,  4096);
  wtrans_kernel<<<dim3(2048 / 32, 2048 / 32), 256, 0, stream>>>(w_o,    wo_t,   2048, 2048);

  // q_c = rmsnorm(h @ w_q_a); q = q_c @ w_q_b (dual f32+bf16 output)
  gemm_bf16_wmma<<<dim3(1536 / GN, 4096 / GM), 256, 0, stream>>>(
      h_bf, wqa_t, q_pre, (__bf16*)nullptr, 4096, 1536, 2048);
  rmsnorm_kernel<<<4096, 256, 0, stream>>>(q_pre, q_a_ln_w, q_c_bf, 1536, 1536, 1536);
  gemm_bf16_wmma<<<dim3(3072 / GN, 4096 / GM), 256, 0, stream>>>(
      q_c_bf, wqb_t, qbuf, q_bf, 4096, 3072, 1536);

  // kv_comb = h @ w_kv_a; kv_c = rmsnorm(kv_comb[:, :512]); kv = kv_c @ w_kv_b
  gemm_bf16_wmma<<<dim3(576 / GN, 4096 / GM), 256, 0, stream>>>(
      h_bf, wkva_t, kv_comb, (__bf16*)nullptr, 4096, 576, 2048);
  rmsnorm_kernel<<<4096, 256, 0, stream>>>(kv_comb, kv_a_ln_w, kv_c_bf, 512, 576, 512);
  gemm_bf16_wmma<<<dim3(4096 / GN, 4096 / GM), 256, 0, stream>>>(
      kv_c_bf, wkvb_t, (float*)nullptr, kv_bf, 4096, 4096, 512);

  // RoPE + V transpose packing.
  rope_q_kernel<<<4096, 512, 0, stream>>>(qbuf, q_bf, positions);
  rope_k_kernel<<<(4096 * 32) / 256, 256, 0, stream>>>(kv_comb, positions, kr_bf);
  vtrans_kernel<<<dim3(2048 / 32, 128 / 32, 32), 256, 0, stream>>>(kv_bf, vt_bf);

  // Attention (flash, causal) -> bf16 [4096, 2048].
  mla_attn_kernel<<<dim3(32, 16, 2), 256, 0, stream>>>(q_bf, kv_bf, kr_bf, vt_bf, attn_bf);

  // out = attn @ w_o (f32 output).
  gemm_bf16_wmma<<<dim3(2048 / GN, 4096 / GM), 256, 0, stream>>>(
      attn_bf, wo_t, out, (__bf16*)nullptr, 4096, 2048, 2048);
}
